// ReferenceDeepseekV3Router_41583873359988
// MI455X (gfx1250) — matrix operand order repaired and checked by reading, not compiled, not run on hardware
//
#include <hip/hip_runtime.h>

#define TOKENS   8192
#define HIDDEN   7168
#define NEXP     256
#define NGROUP   8
#define EPG      32      // experts per group
#define TOPKG    4
#define TOPK     8
#define RSCALE   2.5f

#define MBLK     32      // tokens per block
#define KCH      128     // K chunk staged in LDS
#define LDA      (KCH + 8)   // padded A row (bf16 elems): stride 272B, conflict-free
#define SLD      257     // padded score row (f32 elems)
#define NTHREADS 512     // 16 wave32

typedef __attribute__((ext_vector_type(8)))  float   f32x8;
typedef __attribute__((ext_vector_type(8)))  __bf16  bf16x8;
typedef __attribute__((ext_vector_type(16))) __bf16  bf16x16;

union BFrag {
    uint4   q[2];
    bf16x16 v;
};

union Pack8 {
    bf16x8 v;
    uint4  q;
};

// 8 x f32 -> 8 x bf16 using the hardware converter (clang lowers elementwise
// __bf16 casts to v_cvt*_bf16_f32 on gfx1250).
__device__ __forceinline__ uint4 cvt8(const float4 a, const float4 b) {
    f32x8 f = { a.x, a.y, a.z, a.w, b.x, b.y, b.z, b.w };
    Pack8 p;
    p.v = __builtin_convertvector(f, bf16x8);
    return p.q;
}

// ---------------- kernel 0: pre-convert W (f32 -> bf16) into workspace ----------------
__global__ __launch_bounds__(256)
void convert_w_kernel(const float* __restrict__ W, unsigned short* __restrict__ Wb) {
    const size_t i = ((size_t)blockIdx.x * blockDim.x + threadIdx.x) * 8;
    const float4* src = (const float4*)(W + i);
    float4 a0 = src[0];
    float4 a1 = src[1];
    *(uint4*)(Wb + i) = cvt8(a0, a1);
}

// ---------------- kernel 1: fused router GEMM + sigmoid + grouped top-k ----------------
__global__ __launch_bounds__(NTHREADS)
void deepseek_router_kernel(const float* __restrict__ hs,           // [TOKENS, HIDDEN] f32
                            const unsigned short* __restrict__ Wb,  // [NEXP, HIDDEN] bf16
                            const float* __restrict__ bias,         // [NEXP]
                            float* __restrict__ outIdx,             // [TOKENS, TOPK] (as float)
                            float* __restrict__ outW)               // [TOKENS, TOPK]
{
    __shared__ __align__(16) unsigned short aT[MBLK * LDA];   // bf16 A tile
    __shared__ float sS[MBLK * SLD];                          // sigmoid scores [tok][exp]
    __shared__ float sBias[NEXP];

    const int tid  = threadIdx.x;
    const int lane = tid & 31;
    const int wave = tid >> 5;
    const int tokBase = blockIdx.x * MBLK;
    const int eBase   = wave * 16;

    if (tid < NEXP) sBias[tid] = bias[tid];

    f32x8 acc0 = {0.f,0.f,0.f,0.f,0.f,0.f,0.f,0.f};
    f32x8 acc1 = {0.f,0.f,0.f,0.f,0.f,0.f,0.f,0.f};

    // A staging map: each thread converts 8 f32 -> 8 bf16 (one 16B LDS store)
    const int sRow = tid >> 4;          // 0..31
    const int sCol = (tid & 15) * 8;    // 0..120

    const float* aSrcBase = hs + (size_t)(tokBase + sRow) * HIDDEN + sCol;
    const int laneLo = lane & 15;
    const int bKsel  = (lane >> 4) << 3;  // +0 / +8  K for B fragment
    const int aKsel  = (lane >> 4) << 3;  // +0 / +8  K for A fragment
    const unsigned short* wbRow = Wb + (size_t)(eBase + laneLo) * HIDDEN;

    for (int kc = 0; kc < HIDDEN; kc += KCH) {
        __syncthreads();
        // ---- stage A chunk (f32 -> bf16, hw converter) into LDS ----
        const float4* src = (const float4*)(aSrcBase + kc);
        float4 a0 = src[0];
        float4 a1 = src[1];
        *(uint4*)&aT[sRow * LDA + sCol] = cvt8(a0, a1);
        __syncthreads();

        // ---- 4 WMMA K-steps over the chunk ----
        #pragma unroll
        for (int ks = 0; ks < KCH; ks += 32) {
            // B fragment: bf16 weights straight from L2, no conversion.
            // column = expert eBase+lane%16, K runs {bKsel, bKsel+16}
            const uint4* wp = (const uint4*)(wbRow + kc + ks + bKsel);
            BFrag b;
            b.q[0] = wp[0];
            b.q[1] = wp[2];

            // A fragment, M-tile 0: row = lane%16, K runs {aKsel, aKsel+16}
            BFrag aF;
            aF.q[0] = *(const uint4*)&aT[laneLo * LDA + ks + aKsel];
            aF.q[1] = *(const uint4*)&aT[laneLo * LDA + ks + aKsel + 16];
            acc0 = __builtin_amdgcn_wmma_f32_16x16x32_bf16(
                       false, aF.v, false, b.v, (short)0, acc0, false, false);

            // A fragment, M-tile 1: rows 16..31
            BFrag aG;
            aG.q[0] = *(const uint4*)&aT[(16 + laneLo) * LDA + ks + aKsel];
            aG.q[1] = *(const uint4*)&aT[(16 + laneLo) * LDA + ks + aKsel + 16];
            acc1 = __builtin_amdgcn_wmma_f32_16x16x32_bf16(
                       false, aG.v, false, b.v, (short)0, acc1, false, false);
        }
    }

    __syncthreads();
    // ---- epilogue: sigmoid, scatter scores to LDS per C-matrix layout ----
    {
        const int n   = eBase + laneLo;
        const int mHi = (lane >> 4) * 8;   // VGPR r -> M = r + 8*(lane>=16)
        #pragma unroll
        for (int r = 0; r < 8; ++r) {
            sS[(mHi + r) * SLD + n]      = 1.0f / (1.0f + __expf(-acc0[r]));
            sS[(16 + mHi + r) * SLD + n] = 1.0f / (1.0f + __expf(-acc1[r]));
        }
    }
    __syncthreads();

    // ---- routing: one thread per token ----
    if (tid < MBLK) {
        const int token = tokBase + tid;
        const float* row = &sS[tid * SLD];

        // group score = sum of top-2 biased scores within group
        float gs[NGROUP];
        #pragma unroll
        for (int g = 0; g < NGROUP; ++g) {
            float m1 = -1e30f, m2 = -1e30f;
            for (int j = 0; j < EPG; ++j) {
                int e = g * EPG + j;
                float v = row[e] + sBias[e];
                if (v > m1)      { m2 = m1; m1 = v; }
                else if (v > m2) { m2 = v; }
            }
            gs[g] = m1 + m2;
        }

        // top-4 groups (stable: lowest index wins ties)
        unsigned gmask = 0;
        #pragma unroll
        for (int s = 0; s < TOPKG; ++s) {
            int best = 0; float bv = -1e30f;
            for (int g = 0; g < NGROUP; ++g)
                if (!((gmask >> g) & 1u) && gs[g] > bv) { bv = gs[g]; best = g; }
            gmask |= 1u << best;
        }

        // top-8 experts over masked biased scores (non-selected groups -> 0.0)
        float tv[TOPK]; int ti[TOPK];
        #pragma unroll
        for (int i = 0; i < TOPK; ++i) { tv[i] = -1e30f; ti[i] = 0; }
        for (int e = 0; e < NEXP; ++e) {
            float v = ((gmask >> (e >> 5)) & 1u) ? (row[e] + sBias[e]) : 0.0f;
            if (v > tv[TOPK - 1]) {
                int j = TOPK - 1;
                while (j > 0 && v > tv[j - 1]) {
                    tv[j] = tv[j - 1]; ti[j] = ti[j - 1]; --j;
                }
                tv[j] = v; ti[j] = e;
            }
        }

        // weights from raw sigmoid scores, normalize, scale
        float w[TOPK]; float sum = 0.f;
        #pragma unroll
        for (int i = 0; i < TOPK; ++i) { w[i] = row[ti[i]]; sum += w[i]; }
        const float inv = RSCALE / (sum + 1e-20f);
        #pragma unroll
        for (int i = 0; i < TOPK; ++i) {
            outIdx[token * TOPK + i] = (float)ti[i];
            outW[token * TOPK + i]   = w[i] * inv;
        }
    }
}

extern "C" void kernel_launch(void* const* d_in, const int* in_sizes, int n_in,
                              void* d_out, int out_size, void* d_ws, size_t ws_size,
                              hipStream_t stream) {
    (void)in_sizes; (void)n_in; (void)out_size; (void)ws_size;
    const float* hs   = (const float*)d_in[0];
    const float* W    = (const float*)d_in[1];
    const float* bias = (const float*)d_in[2];
    float* out    = (float*)d_out;
    float* outIdx = out;                       // first TOKENS*TOPK elems
    float* outW   = out + (size_t)TOKENS * TOPK;

    unsigned short* Wb = (unsigned short*)d_ws;  // [NEXP*HIDDEN] bf16, 3.67 MB

    // 1) pre-convert weights to bf16 (3.7 MB, stays L2-resident)
    {
        const size_t elems = (size_t)NEXP * HIDDEN;           // 1,835,008
        const int threads = 256;
        const int blocks  = (int)(elems / (8 * threads));     // 896
        convert_w_kernel<<<blocks, threads, 0, stream>>>(W, Wb);
    }

    // 2) fused router: GEMM (bf16 WMMA) + sigmoid + grouped top-k
    {
        dim3 grid(TOKENS / MBLK);   // 256 blocks
        dim3 block(NTHREADS);       // 16 wave32
        deepseek_router_kernel<<<grid, block, 0, stream>>>(hs, Wb, bias, outIdx, outW);
    }
}
